// MA3D_A_15341623181642
// MI455X (gfx1250) — matrix operand order repaired and checked by reading, not compiled, not run on hardware
//
#include <hip/hip_runtime.h>

typedef __attribute__((ext_vector_type(16))) __bf16 v16bf;
typedef __attribute__((ext_vector_type(4)))  __bf16 v4bf;
typedef __attribute__((ext_vector_type(8)))  float  v8f;

#define BATCH 4
#define SLEN  2048
#define EMB   1024
#define HEADS 16
#define DHEAD 64

__device__ __forceinline__ v8f vzero8() {
    v8f z;
#pragma unroll
    for (int i = 0; i < 8; ++i) z[i] = 0.0f;
    return z;
}

__device__ __forceinline__ v8f wmma_bf16(v16bf a, v16bf b, v8f c) {
    return __builtin_amdgcn_wmma_f32_16x16x32_bf16(
        /*neg_a=*/false, a, /*neg_b=*/false, b,
        /*c_mod=*/(short)0, c, /*reuse_a=*/false, /*reuse_b=*/false);
}

// ---------------------------------------------------------------------------
// f32 -> bf16 conversion (4 elements / thread)
// ---------------------------------------------------------------------------
__global__ void cvt_bf16(const float* __restrict__ in, __bf16* __restrict__ out, int n) {
    int i = (blockIdx.x * blockDim.x + threadIdx.x) * 4;
    if (i < n) {
        float4 f = *(const float4*)(in + i);
        v4bf o;
        o[0] = (__bf16)f.x; o[1] = (__bf16)f.y;
        o[2] = (__bf16)f.z; o[3] = (__bf16)f.w;
        *(v4bf*)(out + i) = o;
    }
}

// ---------------------------------------------------------------------------
// Transpose + convert: in f32 [R][C] -> out bf16 [C][R]. 32x32 LDS tiles,
// coalesced on both global sides. R, C multiples of 32.
// ---------------------------------------------------------------------------
__global__ __launch_bounds__(256) void cvt_t_bf16(const float* __restrict__ in,
                                                  __bf16* __restrict__ out,
                                                  int R, int C) {
    __shared__ float tile[32][33];
    const int r0 = blockIdx.y * 32, c0 = blockIdx.x * 32;
    const int lx = threadIdx.x & 31, ly = threadIdx.x >> 5;   // ly: 0..7
#pragma unroll
    for (int j = 0; j < 4; ++j)
        tile[ly + 8 * j][lx] = in[(size_t)(r0 + ly + 8 * j) * C + c0 + lx];
    __syncthreads();
#pragma unroll
    for (int j = 0; j < 4; ++j)
        out[(size_t)(c0 + ly + 8 * j) * R + r0 + lx] = (__bf16)tile[lx][ly + 8 * j];
}

// ---------------------------------------------------------------------------
// QKV GEMM: X[8192 x 1024] bf16 @ Wt'[3072 x 1024] (N-major) + bias(f32).
// Workgroup tile 128(M) x 64(N); wave tile 32x32 (2x2 WMMA tiles).
// Epilogue scatters q/k into [B,H,S,D], v into [B,H,D,S] (pre-transposed).
// ---------------------------------------------------------------------------
__global__ __launch_bounds__(256) void qkv_gemm(
    const __bf16* __restrict__ X, const __bf16* __restrict__ Wt,
    const float* __restrict__ bias,
    __bf16* __restrict__ qout, __bf16* __restrict__ kout, __bf16* __restrict__ vout)
{
    const int tid  = threadIdx.x;
    const int wave = tid >> 5, lane = tid & 31;
    const int half = lane >> 4, ln = lane & 15;
    const int m0 = blockIdx.y * 128;
    const int n0 = blockIdx.x * 64;
    const int wm = (wave >> 1) * 32;
    const int wn = (wave & 1) * 32;

    __shared__ __bf16 As[128 * 40];    // [m][k], k-contiguous, stride 40
    __shared__ __bf16 Bst[64 * 40];    // [n][k], k-contiguous, stride 40

    v8f acc[2][2];
#pragma unroll
    for (int im = 0; im < 2; ++im)
#pragma unroll
        for (int in = 0; in < 2; ++in) acc[im][in] = vzero8();

    const int a_row = tid >> 1, a_cb = (tid & 1) * 16;
    const int b_row = tid >> 2, b_cb = (tid & 3) * 8;

    for (int kb = 0; kb < 1024; kb += 32) {
        const __bf16* ga = X  + (size_t)(m0 + a_row) * 1024 + kb + a_cb;
        const __bf16* gb = Wt + (size_t)(n0 + b_row) * 1024 + kb + b_cb;
        *(int4*)&As[a_row * 40 + a_cb]      = *(const int4*)ga;
        *(int4*)&As[a_row * 40 + a_cb + 8]  = *(const int4*)(ga + 8);
        *(int4*)&Bst[b_row * 40 + b_cb]     = *(const int4*)gb;
        if (kb + 32 < 1024) {
            __builtin_prefetch(ga + 32, 0, 1);
            __builtin_prefetch(gb + 32, 0, 1);
        }
        __syncthreads();

        v16bf a[2], bf[2];
#pragma unroll
        for (int im = 0; im < 2; ++im)
#pragma unroll
            for (int i = 0; i < 16; ++i) {
                int kk = (half << 4) + i;
                a[im][i] = As[(wm + im * 16 + ln) * 40 + kk];
            }
#pragma unroll
        for (int in = 0; in < 2; ++in)
#pragma unroll
            for (int i = 0; i < 16; ++i)
                bf[in][i] = Bst[(wn + in * 16 + ln) * 40 + (half << 4) + i];

#pragma unroll
        for (int im = 0; im < 2; ++im)
#pragma unroll
            for (int in = 0; in < 2; ++in)
                acc[im][in] = wmma_bf16(a[im], bf[in], acc[im][in]);
        __syncthreads();
    }

#pragma unroll
    for (int in = 0; in < 2; ++in) {
        int col   = n0 + wn + in * 16 + ln;      // 0..3071
        int which = col >> 10;                   // 0=q 1=k 2=v (uniform per tile)
        int e     = col & 1023;
        int hh    = e >> 6, d = e & 63;
        float bv = bias[col];
        if (which == 2) {
            // V stored transposed: [B,H,D,S]
#pragma unroll
            for (int im = 0; im < 2; ++im)
#pragma unroll
                for (int r = 0; r < 8; ++r) {
                    int row = m0 + wm + im * 16 + r + (half << 3);
                    int bb = row >> 11, ss = row & 2047;
                    vout[(((size_t)bb * HEADS + hh) * DHEAD + d) * SLEN + ss] =
                        (__bf16)(acc[im][in][r] + bv);
                }
        } else {
            __bf16* dst = (which == 0) ? qout : kout;
#pragma unroll
            for (int im = 0; im < 2; ++im)
#pragma unroll
                for (int r = 0; r < 8; ++r) {
                    int row = m0 + wm + im * 16 + r + (half << 3);
                    int bb = row >> 11, ss = row & 2047;
                    dst[(((size_t)bb * HEADS + hh) * SLEN + ss) * DHEAD + d] =
                        (__bf16)(acc[im][in][r] + bv);
                }
        }
    }
}

// ---------------------------------------------------------------------------
// Flash attention (causal, online softmax). grid: (S/128, B*H).
// Q,K: [B,H,S,D]; V: [B,H,D,S] (pre-transposed). Each wave owns 16 queries.
// ---------------------------------------------------------------------------
__global__ __launch_bounds__(256) void attn_kernel(
    const __bf16* __restrict__ q, const __bf16* __restrict__ k,
    const __bf16* __restrict__ v, __bf16* __restrict__ o)
{
    const int qblk = blockIdx.x;
    const int bh   = blockIdx.y;
    const int b    = bh / HEADS, h = bh % HEADS;
    const int tid  = threadIdx.x;
    const int wave = tid >> 5, lane = tid & 31;
    const int half = lane >> 4, ln = lane & 15;

    const size_t bh_off = (size_t)bh * SLEN * DHEAD;
    const __bf16* Q  = q + bh_off;
    const __bf16* K  = k + bh_off;
    const __bf16* Vt = v + bh_off;     // [D][S]

    const int q0 = qblk * 128;
    const int qw = q0 + wave * 16;

    __shared__ __bf16 Ks[64 * 72];     // [key][d], d-contiguous
    __shared__ __bf16 Vst[64 * 72];    // [d][key], key-contiguous
    __shared__ __bf16 Ps[8][16 * 72];  // wave-private P tiles [row][key]

    // Q fragments (2 chunks of K=32 over D=64), A-matrix layout
    v16bf qf[2];
#pragma unroll
    for (int c = 0; c < 2; ++c) {
        const __bf16* qrow = Q + (size_t)(qw + ln) * DHEAD + c * 32 + (half << 3);
#pragma unroll
        for (int i = 0; i < 8; ++i) qf[c][i] = qrow[i];
#pragma unroll
        for (int i = 0; i < 8; ++i) qf[c][8 + i] = qrow[16 + i];
    }

    float mrow[8], lsum[8];
    v8f acc[4];
#pragma unroll
    for (int r = 0; r < 8; ++r) { mrow[r] = -1e30f; lsum[r] = 0.0f; }
#pragma unroll
    for (int t = 0; t < 4; ++t) acc[t] = vzero8();

    const int kend   = q0 + 128;
    const int kend_w = qw + 16;

    const int l_row = tid >> 2, l_cb = (tid & 3) * 16;

    for (int kb = 0; kb < kend; kb += 64) {
        const __bf16* kg = K  + (size_t)(kb + l_row) * DHEAD + l_cb;    // keys: rows
        const __bf16* vg = Vt + (size_t)l_row * SLEN + kb + l_cb;       // d: rows
        *(int4*)&Ks[l_row * 72 + l_cb]      = *(const int4*)kg;
        *(int4*)&Ks[l_row * 72 + l_cb + 8]  = *(const int4*)(kg + 8);
        *(int4*)&Vst[l_row * 72 + l_cb]     = *(const int4*)vg;
        *(int4*)&Vst[l_row * 72 + l_cb + 8] = *(const int4*)(vg + 8);
        __syncthreads();

        if (kb < kend_w) {   // wave-uniform -> EXEC all ones inside
            // S = Q K^T for 4 key tiles of 16
            v8f sc[4];
#pragma unroll
            for (int t = 0; t < 4; ++t) {
                v8f s = vzero8();
#pragma unroll
                for (int c = 0; c < 2; ++c) {
                    v16bf bf;
#pragma unroll
                    for (int i = 0; i < 16; ++i)
                        bf[i] = Ks[(t * 16 + ln) * 72 + c * 32 + ((i >> 3) << 4) + (half << 3) + (i & 7)];
                    s = wmma_bf16(qf[c], bf, s);
                }
                sc[t] = s;
            }
            // scale + causal mask + running max
            float mnew[8];
#pragma unroll
            for (int r = 0; r < 8; ++r) mnew[r] = mrow[r];
#pragma unroll
            for (int t = 0; t < 4; ++t) {
                int key = kb + t * 16 + ln;
#pragma unroll
                for (int r = 0; r < 8; ++r) {
                    int qr = qw + r + (half << 3);
                    float sv = sc[t][r] * 0.125f;          // 1/sqrt(64)
                    sv = (key <= qr) ? sv : -1e30f;
                    sc[t][r] = sv;
                    mnew[r] = fmaxf(mnew[r], sv);
                }
            }
#pragma unroll
            for (int r = 0; r < 8; ++r) {
                float m = mnew[r];
#pragma unroll
                for (int off = 1; off < 16; off <<= 1)
                    m = fmaxf(m, __shfl_xor(m, off, 32));
                mnew[r] = m;
            }
            // exponentiate, row sums, rescale accumulators
            float rs[8];
#pragma unroll
            for (int r = 0; r < 8; ++r) rs[r] = 0.0f;
#pragma unroll
            for (int t = 0; t < 4; ++t)
#pragma unroll
                for (int r = 0; r < 8; ++r) {
                    float p = __expf(sc[t][r] - mnew[r]);
                    sc[t][r] = p;
                    rs[r] += p;
                }
#pragma unroll
            for (int r = 0; r < 8; ++r) {
                float s = rs[r];
#pragma unroll
                for (int off = 1; off < 16; off <<= 1)
                    s += __shfl_xor(s, off, 32);
                float corr = __expf(mrow[r] - mnew[r]);
                lsum[r] = lsum[r] * corr + s;
                mrow[r] = mnew[r];
                rs[r] = corr;
            }
#pragma unroll
            for (int t = 0; t < 4; ++t)
#pragma unroll
                for (int r = 0; r < 8; ++r)
                    acc[t][r] *= rs[r];

            // stage P (16 x 64) into wave-private LDS for A-layout re-gather
            __bf16* P = Ps[wave];
#pragma unroll
            for (int t = 0; t < 4; ++t)
#pragma unroll
                for (int r = 0; r < 8; ++r)
                    P[(r + (half << 3)) * 72 + t * 16 + ln] = (__bf16)sc[t][r];

            // O += P @ V   (K-dim = 64 keys, 2 chunks of 32)
#pragma unroll
            for (int c = 0; c < 2; ++c) {
                v16bf af;
#pragma unroll
                for (int i = 0; i < 16; ++i) {
                    int kk = (half << 4) + i;
                    af[i] = P[ln * 72 + c * 32 + kk];
                }
#pragma unroll
                for (int t = 0; t < 4; ++t) {
                    v16bf bf;
#pragma unroll
                    for (int i = 0; i < 16; ++i)
                        bf[i] = Vst[(t * 16 + ln) * 72 + c * 32 + (half << 4) + i];
                    acc[t] = wmma_bf16(af, bf, acc[t]);
                }
            }
        }
        __syncthreads();
    }

    // normalize and emit merged-head output: o[b][s][h*64 + d]  (bf16)
#pragma unroll
    for (int r = 0; r < 8; ++r) lsum[r] = 1.0f / lsum[r];
    __bf16* Ob = o + (size_t)b * SLEN * EMB + (size_t)h * DHEAD;
#pragma unroll
    for (int t = 0; t < 4; ++t)
#pragma unroll
        for (int r = 0; r < 8; ++r) {
            int srow = qw + r + (half << 3);
            Ob[(size_t)srow * EMB + t * 16 + ln] = (__bf16)(acc[t][r] * lsum[r]);
        }
}

// ---------------------------------------------------------------------------
// Output projection: X[8192 x 1024] bf16 @ Wt'[1024 x 1024] (N-major) -> f32
// ---------------------------------------------------------------------------
__global__ __launch_bounds__(256) void proj_gemm(
    const __bf16* __restrict__ X, const __bf16* __restrict__ Wt,
    const float* __restrict__ bias, float* __restrict__ out)
{
    const int tid  = threadIdx.x;
    const int wave = tid >> 5, lane = tid & 31;
    const int half = lane >> 4, ln = lane & 15;
    const int m0 = blockIdx.y * 128;
    const int n0 = blockIdx.x * 64;
    const int wm = (wave >> 1) * 32;
    const int wn = (wave & 1) * 32;

    __shared__ __bf16 As[128 * 40];
    __shared__ __bf16 Bst[64 * 40];

    v8f acc[2][2];
#pragma unroll
    for (int im = 0; im < 2; ++im)
#pragma unroll
        for (int in = 0; in < 2; ++in) acc[im][in] = vzero8();

    const int a_row = tid >> 1, a_cb = (tid & 1) * 16;
    const int b_row = tid >> 2, b_cb = (tid & 3) * 8;

    for (int kb = 0; kb < 1024; kb += 32) {
        const __bf16* ga = X  + (size_t)(m0 + a_row) * 1024 + kb + a_cb;
        const __bf16* gb = Wt + (size_t)(n0 + b_row) * 1024 + kb + b_cb;
        *(int4*)&As[a_row * 40 + a_cb]     = *(const int4*)ga;
        *(int4*)&As[a_row * 40 + a_cb + 8] = *(const int4*)(ga + 8);
        *(int4*)&Bst[b_row * 40 + b_cb]    = *(const int4*)gb;
        if (kb + 32 < 1024) {
            __builtin_prefetch(ga + 32, 0, 1);
            __builtin_prefetch(gb + 32, 0, 1);
        }
        __syncthreads();

        v16bf a[2], bf[2];
#pragma unroll
        for (int im = 0; im < 2; ++im)
#pragma unroll
            for (int i = 0; i < 16; ++i) {
                int kk = (half << 4) + i;
                a[im][i] = As[(wm + im * 16 + ln) * 40 + kk];
            }
#pragma unroll
        for (int in = 0; in < 2; ++in)
#pragma unroll
            for (int i = 0; i < 16; ++i)
                bf[in][i] = Bst[(wn + in * 16 + ln) * 40 + (half << 4) + i];

#pragma unroll
        for (int im = 0; im < 2; ++im)
#pragma unroll
            for (int in = 0; in < 2; ++in)
                acc[im][in] = wmma_bf16(a[im], bf[in], acc[im][in]);
        __syncthreads();
    }

#pragma unroll
    for (int in = 0; in < 2; ++in) {
        int col = n0 + wn + in * 16 + ln;
        float bv = bias[col];
#pragma unroll
        for (int im = 0; im < 2; ++im)
#pragma unroll
            for (int r = 0; r < 8; ++r) {
                int row = m0 + wm + im * 16 + r + (half << 3);
                out[(size_t)row * 1024 + col] = acc[im][in][r] + bv;
            }
    }
}

// ---------------------------------------------------------------------------
extern "C" void kernel_launch(void* const* d_in, const int* in_sizes, int n_in,
                              void* d_out, int out_size, void* d_ws, size_t ws_size,
                              hipStream_t stream) {
    const float* x      = (const float*)d_in[0];
    const float* w_attn = (const float*)d_in[1];
    const float* b_attn = (const float*)d_in[2];
    const float* w_proj = (const float*)d_in[3];
    const float* b_proj = (const float*)d_in[4];
    float* out = (float*)d_out;

    const size_t N_X  = (size_t)BATCH * SLEN * EMB;   // 8,388,608
    const size_t N_WA = (size_t)EMB * 3 * EMB;        // 3,145,728
    const size_t N_WP = (size_t)EMB * EMB;            // 1,048,576

    char* ws = (char*)d_ws;
    size_t off = 0;
    auto carve = [&](size_t bytes) {
        void* p = ws + off;
        off = (off + bytes + 255) & ~(size_t)255;
        return p;
    };
    __bf16* xb   = (__bf16*)carve(N_X * 2);   // reused as attention output
    __bf16* wabT = (__bf16*)carve(N_WA * 2);  // [3072][1024]
    __bf16* wpbT = (__bf16*)carve(N_WP * 2);  // [1024][1024]
    __bf16* qb   = (__bf16*)carve(N_X * 2);   // [B,H,S,D]
    __bf16* kb   = (__bf16*)carve(N_X * 2);   // [B,H,S,D]
    __bf16* vb   = (__bf16*)carve(N_X * 2);   // [B,H,D,S] (transposed)
    __bf16* ob   = xb;
    (void)ws_size; (void)in_sizes; (void)n_in; (void)out_size;

    cvt_bf16<<<(int)(N_X / 4 / 256), 256, 0, stream>>>(x, xb, (int)N_X);
    cvt_t_bf16<<<dim3(3 * EMB / 32, EMB / 32), 256, 0, stream>>>(w_attn, wabT, EMB, 3 * EMB);
    cvt_t_bf16<<<dim3(EMB / 32, EMB / 32), 256, 0, stream>>>(w_proj, wpbT, EMB, EMB);

    qkv_gemm<<<dim3(3 * EMB / 64, BATCH * SLEN / 128), 256, 0, stream>>>(
        xb, wabT, b_attn, qb, kb, vb);

    attn_kernel<<<dim3(SLEN / 128, BATCH * HEADS), 256, 0, stream>>>(qb, kb, vb, ob);

    proj_gemm<<<dim3(EMB / 64, BATCH * SLEN / 128), 256, 0, stream>>>(
        ob, wpbT, b_proj, out);
}
